// SMC_32238024524431
// MI455X (gfx1250) — matrix operand order repaired and checked by reading, not compiled, not run on hardware
//
#include <hip/hip_runtime.h>
#include <hip/hip_bf16.h>

typedef __attribute__((ext_vector_type(16))) __bf16 v16bf;
typedef __attribute__((ext_vector_type(8)))  float  v8f;

union ABfrag {
  uint4 u4[2];
  v16bf v;
};

__device__ __forceinline__ unsigned int f2bf(float f) {
  union { float f; unsigned int u; } cvt;
  cvt.f = f;
  unsigned int u = cvt.u;
  unsigned int r = u + 0x7FFFu + ((u >> 16) & 1u);   // round-to-nearest-even
  return r >> 16;
}

__device__ __forceinline__ unsigned int pk2bf(float lo, float hi) {
  return f2bf(lo) | (f2bf(hi) << 16);
}

// ---------------------------------------------------------------------------
// Kernel 1: fold per-channel scales into weights, convert to bf16, and pack
// into the CDNA5 16-bit A-matrix (16x32) VGPR layout (ISA 7.12.2) so the main
// kernel loads each A fragment as two coalesced b128 loads per lane.
//
// ws layout: [tap(9)][kchunk(4)][mtile(8)][lane(32)][elem(16)] (ushort)
//   elem e -> VGPR j = e/2, half = e&1
//   lanes 0-15 : M = lane,    K = j<4 ? 2j+half     : 16+2(j-4)+half
//   lanes 16-31: M = lane-16, K = j<4 ? 8+2j+half   : 24+2(j-4)+half
// ---------------------------------------------------------------------------
__global__ __launch_bounds__(256) void smc_pack_weights(
    const float* __restrict__ w, const float* __restrict__ scales,
    unsigned short* __restrict__ wpk) {
  int gid = blockIdx.x * 256 + threadIdx.x;      // 0 .. 147455
  int e  = gid & 15;
  int l  = (gid >> 4) & 31;
  int mt = (gid >> 9) & 7;
  int q  = (gid >> 12) & 3;
  int t  = gid >> 14;                            // 0..8
  int kh = t / 3, kw = t % 3;
  int j = e >> 1, half = e & 1;
  int o = mt * 16 + (l & 15);
  int K;
  if (l < 16) K = (j < 4) ? (2 * j + half)     : (16 + 2 * (j - 4) + half);
  else        K = (j < 4) ? (8 + 2 * j + half) : (24 + 2 * (j - 4) + half);
  int c = q * 32 + K;
  float v = w[((o * 128 + c) * 3 + kh) * 3 + kw] * scales[c];
  wpk[gid] = (unsigned short)f2bf(v);
}

// ---------------------------------------------------------------------------
// Kernel 2: implicit-GEMM conv, 4-row x 32-col output tile per WG,
// double-buffered LDS, register staging with pure shift/mask index math
// (interior 6x32x32 = 12 channel-pairs/thread; 2-col halo = 192 pairs).
// Chunk q+1's HBM loads are in flight during chunk q's 72 WMMAs.
// ---------------------------------------------------------------------------
__global__ __launch_bounds__(256) void smc_conv_wmma(
    const float* __restrict__ x, const unsigned short* __restrict__ wpk,
    const float* __restrict__ gain, const float* __restrict__ bias,
    float* __restrict__ out) {
  __shared__ unsigned short xs[2][6528];         // 2 x 6x34x32 bf16 tiles

  const int tid  = threadIdx.x;
  const int lane = tid & 31;
  const int wv   = tid >> 5;
  const int bid  = blockIdx.x;
  const int wblk = bid & 7;                      // 8 col blocks of 32
  const int hblk = (bid >> 3) & 63;              // 64 row blocks of 4
  const int b    = bid >> 9;
  const int h0   = hblk * 4;
  const int w0   = wblk * 32;

  v8f acc[8] = {};                               // 8 cout tiles x 16x16 f32

  const int pix = lane & 15;                     // pixel column within N tile
  const int ks  = (lane < 16) ? 0 : 8;           // K octet base for B fragment
  const int rr  = wv >> 1;                       // output row within tile
  const int cs  = (wv & 1) * 16;                 // column half within tile

  float rbuf[24];                                // interior: 12 channel pairs
  float hbuf[2];                                 // halo pair (tid < 192)

  // issue: start global loads for K chunk q; clamped-row loads + select
  // (no exec-mask branches, 32-bit offsets -> saddr+voffset addressing)
  auto issue = [&](int q) {
    const int cbase = b * 128 + q * 32;
    #pragma unroll
    for (int i = 0; i < 12; ++i) {
      int pidx = tid + 256 * i;                  // 0..3071
      int ww = pidx & 31;
      int cp = (pidx >> 5) & 15;
      int hh = pidx >> 9;                        // 0..5
      int gh = h0 - 1 + hh;
      int ghc = min(max(gh, 0), 255);            // v_med3
      bool okh = (unsigned)gh < 256u;
      int off = ((cbase + 2 * cp) * 256 + ghc) * 256 + (w0 + ww);
      float v0 = x[off];
      float v1 = x[off + 65536];
      rbuf[2 * i]     = okh ? v0 : 0.0f;
      rbuf[2 * i + 1] = okh ? v1 : 0.0f;
    }
    if (tid < 192) {                             // halo cols ww = -1, +32
      int col = tid & 1;
      int cp  = (tid >> 1) & 15;
      int hh  = tid >> 5;                        // 0..5
      int gh  = h0 - 1 + hh;
      int gw  = w0 - 1 + col * 33;
      int ghc = min(max(gh, 0), 255);
      int gwc = min(max(gw, 0), 255);
      bool ok = ((unsigned)gh < 256u) && ((unsigned)gw < 256u);
      int off = ((cbase + 2 * cp) * 256 + ghc) * 256 + gwc;
      float v0 = x[off];
      float v1 = x[off + 65536];
      hbuf[0] = ok ? v0 : 0.0f;
      hbuf[1] = ok ? v1 : 0.0f;
    }
  };

  // commit: pack bf16 pairs, one b32 LDS store per pair
  auto commit = [&](int qb) {
    unsigned int* xd = (unsigned int*)xs[qb];
    #pragma unroll
    for (int i = 0; i < 12; ++i) {
      int pidx = tid + 256 * i;
      int ww = pidx & 31;
      int cp = (pidx >> 5) & 15;
      int hh = pidx >> 9;
      xd[(hh * 34 + ww + 1) * 16 + cp] = pk2bf(rbuf[2 * i], rbuf[2 * i + 1]);
    }
    if (tid < 192) {
      int col = tid & 1;
      int cp  = (tid >> 1) & 15;
      int hh  = tid >> 5;
      xd[(hh * 34 + col * 33) * 16 + cp] = pk2bf(hbuf[0], hbuf[1]);
    }
  };

  // compute: 9 taps x 8 cout tiles of bf16 WMMA from LDS buffer qb
  auto compute = [&](int q, int qb) {
    #pragma unroll
    for (int t = 0; t < 9; ++t) {
      const int kh = t / 3, kw = t % 3;
      ABfrag bf;                                 // pixels (32x16), 2x b128
      const int off = ((rr + kh) * 34 + (cs + pix + kw)) * 32 + ks;
      bf.u4[0] = *(const uint4*)(&xs[qb][off]);
      bf.u4[1] = *(const uint4*)(&xs[qb][off + 16]);
      #pragma unroll
      for (int mt = 0; mt < 8; ++mt) {
        ABfrag af;                               // pre-packed weights
        const uint4* wp = (const uint4*)(
            wpk + (((t * 4 + q) * 8 + mt) * 512 + lane * 16));
        af.u4[0] = wp[0];
        af.u4[1] = wp[1];
        acc[mt] = __builtin_amdgcn_wmma_f32_16x16x32_bf16(
            false, af.v, false, bf.v, (short)0, acc[mt], false, false);
      }
    }
  };

  // ---- software pipeline over the 4 K chunks ----
  issue(0);
  commit(0);
  __syncthreads();
  #pragma unroll 1
  for (int q = 0; q < 4; ++q) {
    if (q < 3) issue(q + 1);                     // HBM loads in flight...
    compute(q, q & 1);                           // ...during these WMMAs
    if (q < 3) commit((q + 1) & 1);              // other buffer: race-free
    __syncthreads();
  }

  // ---- epilogue: y = acc*gain + bias, coalesced non-temporal stores ----
  const float g  = gain[0];
  const int p    = w0 + cs + pix;
  const int h    = h0 + rr;
  const int mrow = (lane < 16) ? 0 : 8;
  #pragma unroll
  for (int mt = 0; mt < 8; ++mt) {
    const float4* bp = (const float4*)&bias[mt * 16 + mrow];
    float4 b0 = bp[0];
    float4 b1 = bp[1];
    const float bv[8] = {b0.x, b0.y, b0.z, b0.w, b1.x, b1.y, b1.z, b1.w};
    #pragma unroll
    for (int r = 0; r < 8; ++r) {
      int o = mt * 16 + mrow + r;
      __builtin_nontemporal_store(
          acc[mt][r] * g + bv[r],
          &out[((b * 128 + o) * 256 + h) * 256 + p]);
    }
  }
}

extern "C" void kernel_launch(void* const* d_in, const int* in_sizes, int n_in,
                              void* d_out, int out_size, void* d_ws, size_t ws_size,
                              hipStream_t stream) {
  const float* x      = (const float*)d_in[0];
  const float* w      = (const float*)d_in[1];
  const float* scales = (const float*)d_in[2];
  const float* gain   = (const float*)d_in[3];
  const float* bias   = (const float*)d_in[4];
  float* out          = (float*)d_out;
  unsigned short* wpk = (unsigned short*)d_ws;   // 147456 ushorts = 288 KB

  // 9*4*8*32*16 = 147456 packed weight elements
  smc_pack_weights<<<576, 256, 0, stream>>>(w, scales, wpk);
  // 8 batches * 64 row-blocks * 8 col-blocks
  smc_conv_wmma<<<4096, 256, 0, stream>>>(x, wpk, gain, bias, out);
}
